// FixedSegmentDictDMPNet_17506286699079
// MI455X (gfx1250) — matrix operand order repaired and checked by reading, not compiled, not run on hardware
//
#include <hip/hip_runtime.h>
#include <hip/hip_bf16.h>

// ---------------------------------------------------------------------------
// CDNA5 (gfx1250): implicit-GEMM convs + FC layers on v_wmma_f32_16x16x32_bf16.
// All GEMM operands are bf16, contiguous along K, loaded as aligned b128s:
//   - conv weights packed [Cout][9*Cin] with k = (ky*3+kx)*Cin + ci
//   - activations NHWC (conv out f32, pool out bf16)
//   - FC weights transpose-packed [N][K] bf16 (fc1 includes NCHW->NHWC perm)
//   - conv1 (Cin=3) runs as a pure GEMM on a materialized im2col [N][32]
// Each wave computes a 64(M) x 32(N) slab: 8 independent WMMA accumulators,
// so every A fragment feeds 2 WMMAs and hazard NOPs are fully hidden.
// ---------------------------------------------------------------------------

typedef __bf16 bf16;
typedef __attribute__((ext_vector_type(16))) __bf16 v16bf;
typedef __attribute__((ext_vector_type(8)))  __bf16 v8bf;
typedef __attribute__((ext_vector_type(8)))  float  v8f;

#define WAVES_PER_BLOCK 4
#define BLOCK_THREADS   (WAVES_PER_BLOCK * 32)
#define MTILES          4      // 4 x 16 = 64 output rows per wave
#define NTILES          2      // 2 x 16 = 32 output cols per wave

// Load a 16x32 A fragment row-slice: elements {0..7,16..23} + 8*hi (bf16).
__device__ __forceinline__ v16bf load_a_frag(const bf16* __restrict__ wrow, int hi) {
  v8bf lo = *(const v8bf*)(wrow + 8 * hi);        // 16B aligned
  v8bf hh = *(const v8bf*)(wrow + 16 + 8 * hi);   // 16B aligned
  return __builtin_shufflevector(lo, hh, 0, 1, 2, 3, 4, 5, 6, 7,
                                 8, 9, 10, 11, 12, 13, 14, 15);
}

// B fragment: elements {0..7,16..23} + 8*hi (bf16), matching A; 16B aligned.
__device__ __forceinline__ v16bf load_b_frag(const bf16* __restrict__ rowp, int hi) {
  v8bf lo = *(const v8bf*)(rowp + 8 * hi);
  v8bf hh = *(const v8bf*)(rowp + 16 + 8 * hi);
  return __builtin_shufflevector(lo, hh, 0, 1, 2, 3, 4, 5, 6, 7,
                                 8, 9, 10, 11, 12, 13, 14, 15);
}

// ===========================================================================
// Prepack kernels (one-time, memory-bound)
// ===========================================================================

// conv1 weights: OIHW f32 -> [64][32] bf16, k = ci*9 + r (matches im2col1).
__global__ void pack_w1(const float* __restrict__ w, bf16* __restrict__ wp) {
  int i = blockIdx.x * blockDim.x + threadIdx.x;     // 64*32
  if (i >= 64 * 32) return;
  int o = i >> 5, k = i & 31;
  float v = (k < 27) ? w[o * 27 + k] : 0.0f;
  wp[i] = (bf16)v;
}

// conv2-4 weights: OIHW f32 -> [Cout][9*Cin] bf16, k = r*Cin + ci (Cin pow2).
__global__ void pack_w_nhwc(const float* __restrict__ w, bf16* __restrict__ wp,
                            int Cout, int Cin, int csh) {
  int K = 9 << csh;
  int i = blockIdx.x * blockDim.x + threadIdx.x;
  if (i >= Cout * K) return;
  int m = i / K;
  int k = i - m * K;
  int r  = k >> csh;
  int ci = k & (Cin - 1);
  wp[i] = (bf16)w[(long)(m * Cin + ci) * 9 + r];
}

// conv1 im2col: x NCHW f32 (128,3,100,100) -> [N=128*98*98][32] bf16.
__global__ void im2col1_pack(const float* __restrict__ x, bf16* __restrict__ im,
                             int total) {
  int i = blockIdx.x * blockDim.x + threadIdx.x;
  if (i >= total) return;
  int k = i & 31;
  int n = i >> 5;
  int wo = n % 98;
  int t  = n / 98;
  int ho = t % 98;
  int b  = t / 98;
  float v = 0.0f;
  if (k < 27) {
    int ci = k / 9;
    int r  = k - 9 * ci;
    int ky = r / 3;
    int kx = r - 3 * ky;
    v = x[((long)(b * 3 + ci) * 100 + ho + ky) * 100 + wo + kx];
  }
  im[i] = (bf16)v;
}

// fc1 weights: fw1 [8192][2048] f32 -> BT1 [2048][8192] bf16 with the
// NCHW->NHWC feature permutation folded in: knew=(h*4+w)*512+c, f=c*16+pos.
__global__ void pack_bt1(const float* __restrict__ w, bf16* __restrict__ bt) {
  int i = blockIdx.x * blockDim.x + threadIdx.x;     // 2048*8192
  if (i >= 2048 * 8192) return;
  int n    = i >> 13;
  int knew = i & 8191;
  int pos  = knew >> 9;
  int c    = knew & 511;
  int f    = c * 16 + pos;
  bt[i] = (bf16)w[(long)f * 2048 + n];
}

// Generic FC weight transpose: W [K][N] f32 -> BT [Npad][K] bf16 (zero pad).
__global__ void pack_bt(const float* __restrict__ w, bf16* __restrict__ bt,
                        int K, int N, int Npad) {
  int i = blockIdx.x * blockDim.x + threadIdx.x;
  if (i >= Npad * K) return;
  int n = i / K;
  int k = i - n * K;
  float v = (n < N) ? w[(long)k * N + n] : 0.0f;
  bt[i] = (bf16)v;
}

// ===========================================================================
// conv1 as pure GEMM: D[n][m] = im[n][0:32] . wp[m][0:32], single K step.
// out: NHWC f32 [N][64] + bias. Wave covers 2 N-tiles x 4 M-tiles.
// ===========================================================================
__global__ void conv1_gemm(const bf16* __restrict__ im,   // [N][32]
                           const bf16* __restrict__ wpk,  // [64][32]
                           const float* __restrict__ bias,
                           float* __restrict__ out) {     // [N][64]
  const int lane = threadIdx.x & 31;
  const int wave = threadIdx.x >> 5;
  const int hi   = lane >> 4;
  const int ml   = lane & 15;

  int nn[NTILES];
  v16bf bfr[NTILES];
#pragma unroll
  for (int j = 0; j < NTILES; ++j) {
    nn[j] = ((blockIdx.x * WAVES_PER_BLOCK + wave) * NTILES + j) * 16 + ml;
    bfr[j] = load_b_frag(im + (long)nn[j] * 32, hi);
  }

  v8f acc[MTILES][NTILES];
#pragma unroll
  for (int mt = 0; mt < MTILES; ++mt) {
    v16bf a = load_a_frag(wpk + (long)(mt * 16 + ml) * 32, hi);
#pragma unroll
    for (int j = 0; j < NTILES; ++j) {
      v8f z = (v8f){0.f, 0.f, 0.f, 0.f, 0.f, 0.f, 0.f, 0.f};
      acc[mt][j] = __builtin_amdgcn_wmma_f32_16x16x32_bf16(
          false, a, false, bfr[j], (short)0, z, false, false);
    }
  }

#pragma unroll
  for (int j = 0; j < NTILES; ++j) {
    float* outp = out + (long)nn[j] * 64;
#pragma unroll
    for (int mt = 0; mt < MTILES; ++mt)
#pragma unroll
      for (int r = 0; r < 8; ++r) {
        int m = mt * 16 + r + 8 * hi;
        outp[m] = acc[mt][j][r] + bias[m];
      }
  }
}

// ===========================================================================
// conv2-4: NHWC implicit GEMM. in: [B,Hin,Win,Cin] bf16 (Cin = 2^csh >= 64),
// weights [Cout][9*Cin] bf16 (k = r*Cin+ci), out [B,Hout,Wout,Cout] f32+bias.
// Wave covers 2 N-tiles (output pixels) x 4 M-tiles (channels).
// ===========================================================================
__global__ void conv3x3_nhwc_wmma(const bf16* __restrict__ in,
                                  const bf16* __restrict__ wpk,
                                  const float* __restrict__ bias,
                                  float* __restrict__ out,
                                  int csh, int Hin, int Win,
                                  int Cout, int Hout, int Wout) {
  __shared__ int srow[9];
  if (threadIdx.x < 9) {
    int r = threadIdx.x, ky = r / 3, kx = r - 3 * (r / 3);
    srow[r] = (ky * Win + kx) << csh;
  }
  __syncthreads();

  const int lane = threadIdx.x & 31;
  const int wave = threadIdx.x >> 5;
  const int hi   = lane >> 4;
  const int ml   = lane & 15;
  const int HW   = Hout * Wout;

  int nn[NTILES];
  const bf16* pixbase[NTILES];
#pragma unroll
  for (int j = 0; j < NTILES; ++j) {
    int n  = ((blockIdx.x * WAVES_PER_BLOCK + wave) * NTILES + j) * 16 + ml;
    int b  = n / HW;
    int hw = n - b * HW;
    int ho = hw / Wout;
    int wo = hw - ho * Wout;
    nn[j] = n;
    pixbase[j] = in + ((long)((b * Hin + ho) * Win + wo) << csh);
  }

  const int K     = 9 << csh;
  const int cmask = (1 << csh) - 1;
  const int mbase = blockIdx.y * (MTILES * 16);

  v8f acc[MTILES][NTILES];
#pragma unroll
  for (int mt = 0; mt < MTILES; ++mt)
#pragma unroll
    for (int j = 0; j < NTILES; ++j)
      acc[mt][j] = (v8f){0.f, 0.f, 0.f, 0.f, 0.f, 0.f, 0.f, 0.f};

  for (int k0 = 0; k0 < K; k0 += 32) {
    const int r  = k0 >> csh;
    const int cb = k0 & cmask;

    // Prefetch the next (ky,kx) activation row: a strided jump the linear
    // streamer won't anticipate (emits global_prefetch_b8).
    if (cb == 0 && r < 8) {
#pragma unroll
      for (int j = 0; j < NTILES; ++j)
        __builtin_prefetch(pixbase[j] + srow[r + 1] + 16 * hi, 0, 3);
    }

    v16bf bfr[NTILES];
#pragma unroll
    for (int j = 0; j < NTILES; ++j)
      bfr[j] = load_b_frag(pixbase[j] + srow[r] + cb, hi);

#pragma unroll
    for (int mt = 0; mt < MTILES; ++mt) {
      v16bf a = load_a_frag(wpk + (long)(mbase + mt * 16 + ml) * K + k0, hi);
#pragma unroll
      for (int j = 0; j < NTILES; ++j)
        acc[mt][j] = __builtin_amdgcn_wmma_f32_16x16x32_bf16(
            false, a, false, bfr[j], (short)0, acc[mt][j], false, false);
    }
  }

#pragma unroll
  for (int j = 0; j < NTILES; ++j) {
    float* outp = out + (long)nn[j] * Cout;
#pragma unroll
    for (int mt = 0; mt < MTILES; ++mt)
#pragma unroll
      for (int r = 0; r < 8; ++r) {
        int m = mbase + mt * 16 + r + 8 * hi;
        outp[m] = acc[mt][j][r] + bias[m];
      }
  }
}

// ===========================================================================
// 2x2 max pool + ReLU, NHWC f32 -> NHWC bf16. C = 2^csh.
// ===========================================================================
__global__ void pool2_relu_nhwc(const float* __restrict__ in,
                                bf16* __restrict__ out,
                                int Hin, int Win, int csh, int total) {
  int i = blockIdx.x * blockDim.x + threadIdx.x;
  if (i >= total) return;
  int C = 1 << csh;
  int c = i & (C - 1);
  int j = i >> csh;                 // (b*Ho + ho)*Wo + wo
  int Wo = Win >> 1, Ho = Hin >> 1;
  int wo = j % Wo;
  int t  = j / Wo;
  int ho = t % Ho;
  int b  = t / Ho;
  const float* p = in + (((long)(b * Hin + 2 * ho) * Win + 2 * wo) << csh) + c;
  long rs = (long)Win << csh;
  float m0 = fmaxf(p[0], p[C]);
  float m1 = fmaxf(p[rs], p[rs + C]);
  out[i] = (bf16)fmaxf(fmaxf(m0, m1), 0.0f);
}

// ===========================================================================
// FC GEMM: out = act( A[M][K](bf16) @ BT[Npad][K](bf16)^T + bias ).
// act==1: tanh, store bf16 to outH[m*Nstore+n]; act==0: f32 to outF.
// BT is zero-padded so every wave's 2 N-tiles are loadable; stores guarded.
// ===========================================================================
__global__ void fc_wmma(const bf16* __restrict__ A,
                        const bf16* __restrict__ BT,
                        const float* __restrict__ bias,
                        bf16* __restrict__ outH, float* __restrict__ outF,
                        int K, int Nstore, int act) {
  const int lane = threadIdx.x & 31;
  const int wave = threadIdx.x >> 5;
  const int hi   = lane >> 4;
  const int ml   = lane & 15;
  const int mbase = blockIdx.y * (MTILES * 16);

  int nn[NTILES];
  const bf16* brow[NTILES];
#pragma unroll
  for (int j = 0; j < NTILES; ++j) {
    nn[j] = ((blockIdx.x * WAVES_PER_BLOCK + wave) * NTILES + j) * 16 + ml;
    brow[j] = BT + (long)nn[j] * K;
  }

  v8f acc[MTILES][NTILES];
#pragma unroll
  for (int mt = 0; mt < MTILES; ++mt)
#pragma unroll
    for (int j = 0; j < NTILES; ++j)
      acc[mt][j] = (v8f){0.f, 0.f, 0.f, 0.f, 0.f, 0.f, 0.f, 0.f};

  for (int k0 = 0; k0 < K; k0 += 32) {
    v16bf bfr[NTILES];
#pragma unroll
    for (int j = 0; j < NTILES; ++j)
      bfr[j] = load_b_frag(brow[j] + k0, hi);

#pragma unroll
    for (int mt = 0; mt < MTILES; ++mt) {
      v16bf a = load_a_frag(A + (long)(mbase + mt * 16 + ml) * K + k0, hi);
#pragma unroll
      for (int j = 0; j < NTILES; ++j)
        acc[mt][j] = __builtin_amdgcn_wmma_f32_16x16x32_bf16(
            false, a, false, bfr[j], (short)0, acc[mt][j], false, false);
    }
  }

#pragma unroll
  for (int j = 0; j < NTILES; ++j) {
    if (nn[j] < Nstore) {
      float bs = bias[nn[j]];
#pragma unroll
      for (int mt = 0; mt < MTILES; ++mt)
#pragma unroll
        for (int r = 0; r < 8; ++r) {
          int m = mbase + mt * 16 + r + 8 * hi;
          float val = acc[mt][j][r] + bs;
          if (act == 1) outH[(long)m * Nstore + nn[j]] = (bf16)tanhf(val);
          else          outF[(long)m * Nstore + nn[j]] = val;
        }
    }
  }
}

// ===========================================================================
// Trajectory decode. One block per batch element.
// ===========================================================================
__global__ void traj_decode(const float* __restrict__ p,
                            const float* __restrict__ dict,
                            float* __restrict__ out) {
  const int b = blockIdx.x;
  __shared__ int   sidx[20];
  __shared__ float sscale[20][2];
  __shared__ float sstart[20][2];

  const float* pb = p + (long)b * 62;
  if (threadIdx.x == 0) {
    float cx = pb[0], cy = pb[1];
    for (int s = 0; s < 20; ++s) {
      float f  = pb[2 + 3 * s];
      int idx  = (int)fminf(fmaxf(rintf(f), 0.0f), 999.0f);
      float sx = pb[2 + 3 * s + 1];
      float sy = pb[2 + 3 * s + 2];
      sidx[s] = idx;
      sscale[s][0] = sx; sscale[s][1] = sy;
      sstart[s][0] = cx; sstart[s][1] = cy;      // exclusive prefix of offsets
      cx += dict[((long)idx * 200 + 199) * 2 + 0] * sx;
      cy += dict[((long)idx * 200 + 199) * 2 + 1] * sy;
    }
  }
  __syncthreads();

  for (int e = threadIdx.x; e < 20 * 200 * 2; e += blockDim.x) {
    int dim = e & 1;
    int sl  = e >> 1;
    int s   = sl / 200;
    int l   = sl - s * 200;
    out[(long)b * 8000 + e] =
        dict[((long)sidx[s] * 200 + l) * 2 + dim] * sscale[s][dim] + sstart[s][dim];
  }
}

// ===========================================================================
// Host-side orchestration.
// ===========================================================================
static inline size_t ws_take(size_t& off, size_t bytes) {
  off = (off + 255) & ~(size_t)255;
  size_t r = off;
  off += bytes;
  return r;
}

extern "C" void kernel_launch(void* const* d_in, const int* in_sizes, int n_in,
                              void* d_out, int out_size, void* d_ws, size_t ws_size,
                              hipStream_t stream) {
  (void)in_sizes; (void)n_in; (void)out_size; (void)ws_size;

  const float* x    = (const float*)d_in[0];
  const float* w1   = (const float*)d_in[1];
  const float* b1   = (const float*)d_in[2];
  const float* w2   = (const float*)d_in[3];
  const float* b2   = (const float*)d_in[4];
  const float* w3   = (const float*)d_in[5];
  const float* b3   = (const float*)d_in[6];
  const float* w4   = (const float*)d_in[7];
  const float* b4   = (const float*)d_in[8];
  const float* fw1  = (const float*)d_in[9];
  const float* fb1  = (const float*)d_in[10];
  const float* fw2  = (const float*)d_in[11];
  const float* fb2  = (const float*)d_in[12];
  const float* fw3  = (const float*)d_in[13];
  const float* fb3  = (const float*)d_in[14];
  const float* dict = (const float*)d_in[15];
  float* outp = (float*)d_out;

  char* ws = (char*)d_ws;
  size_t off = 0;
  float* bufA  = (float*)(ws + ws_take(off, (size_t)128 * 98 * 98 * 64 * 4)); // conv outs f32 NHWC
  bf16*  bufB  = (bf16*) (ws + ws_take(off, (size_t)128 * 49 * 49 * 64 * 2)); // pool outs bf16 NHWC
  bf16*  im1   = (bf16*) (ws + ws_take(off, (size_t)128 * 98 * 98 * 32 * 2)); // conv1 im2col
  bf16*  wp1   = (bf16*) (ws + ws_take(off, (size_t)64 * 32 * 2));
  bf16*  wp2   = (bf16*) (ws + ws_take(off, (size_t)128 * 576 * 2));
  bf16*  wp3   = (bf16*) (ws + ws_take(off, (size_t)256 * 1152 * 2));
  bf16*  wp4   = (bf16*) (ws + ws_take(off, (size_t)512 * 2304 * 2));
  bf16*  bt1   = (bf16*) (ws + ws_take(off, (size_t)2048 * 8192 * 2));
  bf16*  bt2   = (bf16*) (ws + ws_take(off, (size_t)2048 * 2048 * 2));
  bf16*  bt3   = (bf16*) (ws + ws_take(off, (size_t)128 * 2048 * 2));
  bf16*  fc1o  = (bf16*) (ws + ws_take(off, (size_t)128 * 2048 * 2));
  bf16*  fc2o  = (bf16*) (ws + ws_take(off, (size_t)128 * 2048 * 2));
  float* pbuf  = (float*)(ws + ws_take(off, (size_t)128 * 62 * 4));

  // ---- prepacks ----
  pack_w1<<<(64 * 32 + 255) / 256, 256, 0, stream>>>(w1, wp1);
  pack_w_nhwc<<<(128 * 576 + 255) / 256, 256, 0, stream>>>(w2, wp2, 128, 64, 6);
  pack_w_nhwc<<<(256 * 1152 + 255) / 256, 256, 0, stream>>>(w3, wp3, 256, 128, 7);
  pack_w_nhwc<<<(512 * 2304 + 255) / 256, 256, 0, stream>>>(w4, wp4, 512, 256, 8);
  {
    int tot = 128 * 98 * 98 * 32;
    im2col1_pack<<<(tot + 255) / 256, 256, 0, stream>>>(x, im1, tot);
  }
  pack_bt1<<<(2048 * 8192 + 255) / 256, 256, 0, stream>>>(fw1, bt1);
  pack_bt<<<(2048 * 2048 + 255) / 256, 256, 0, stream>>>(fw2, bt2, 2048, 2048, 2048);
  pack_bt<<<(128 * 2048 + 255) / 256, 256, 0, stream>>>(fw3, bt3, 2048, 62, 128);

  const dim3 blk(BLOCK_THREADS);
  const int WT = WAVES_PER_BLOCK * NTILES;   // N-tiles per block = 8

  // ---- conv1 (GEMM over im2col): N = 128*98*98, K = 32, M = 64 ----
  {
    int ntiles = (128 * 98 * 98) / 16;   // 76832
    conv1_gemm<<<dim3(ntiles / WT, 1), blk, 0, stream>>>(im1, wp1, b1, bufA);
    int tot = 128 * 49 * 49 * 64;
    pool2_relu_nhwc<<<(tot + 255) / 256, 256, 0, stream>>>(bufA, bufB, 98, 98, 6, tot);
  }
  // ---- conv2: (128,49,49,64) -> (128,47,47,128) ----
  {
    int ntiles = (128 * 47 * 47) / 16;   // 17672
    conv3x3_nhwc_wmma<<<dim3(ntiles / WT, 2), blk, 0, stream>>>(
        bufB, wp2, b2, bufA, 6, 49, 49, 128, 47, 47);
    int tot = 128 * 23 * 23 * 128;
    pool2_relu_nhwc<<<(tot + 255) / 256, 256, 0, stream>>>(bufA, bufB, 47, 47, 7, tot);
  }
  // ---- conv3: (128,23,23,128) -> (128,21,21,256) ----
  {
    int ntiles = (128 * 21 * 21) / 16;   // 3528
    conv3x3_nhwc_wmma<<<dim3(ntiles / WT, 4), blk, 0, stream>>>(
        bufB, wp3, b3, bufA, 7, 23, 23, 256, 21, 21);
    int tot = 128 * 10 * 10 * 256;
    pool2_relu_nhwc<<<(tot + 255) / 256, 256, 0, stream>>>(bufA, bufB, 21, 21, 8, tot);
  }
  // ---- conv4: (128,10,10,256) -> (128,8,8,512) ----
  {
    int ntiles = (128 * 8 * 8) / 16;     // 512
    conv3x3_nhwc_wmma<<<dim3(ntiles / WT, 8), blk, 0, stream>>>(
        bufB, wp4, b4, bufA, 8, 10, 10, 512, 8, 8);
    int tot = 128 * 4 * 4 * 512;
    pool2_relu_nhwc<<<(tot + 255) / 256, 256, 0, stream>>>(bufA, bufB, 8, 8, 9, tot);
  }

  // bufB = [128][8192] bf16, NHWC-flatten order (matches BT1's K permutation)

  fc_wmma<<<dim3(2048 / 16 / WT, 2), blk, 0, stream>>>(bufB, bt1, fb1, fc1o,
                                                       nullptr, 8192, 2048, 1);
  fc_wmma<<<dim3(2048 / 16 / WT, 2), blk, 0, stream>>>(fc1o, bt2, fb2, fc2o,
                                                       nullptr, 2048, 2048, 1);
  fc_wmma<<<dim3(1, 2), blk, 0, stream>>>(fc2o, bt3, fb3, nullptr, pbuf,
                                          2048, 62, 0);

  traj_decode<<<128, 256, 0, stream>>>(pbuf, dict, outp);
}
